// EUpdateFormer_60825326846008
// MI455X (gfx1250) — matrix operand order repaired and checked by reading, compiled, not run on hardware
//
#include <hip/hip_runtime.h>
#include <math.h>

// ======================= CDNA5 WMMA types =======================
typedef __attribute__((ext_vector_type(16))) __bf16 v16bf;
typedef __attribute__((ext_vector_type(8)))  float  v8f;

static constexpr int NROW = 256;           // B*N tracks
static constexpr int TLEN = 50;            // sequence length
static constexpr int TOKS = NROW * TLEN;   // 12800 tokens
static constexpr int DMOD = 384;
static constexpr int DINN = 768;           // mLSTM inner dim
static constexpr int DUP  = 1536;
static constexpr int NH   = 4;
static constexpr int DHM  = 192;           // mLSTM head dim
static constexpr int DHS  = 96;            // sLSTM head dim
static constexpr int BH   = NROW * NH;     // 1024 row-heads
static constexpr int QKLD = 64;            // padded ld for 50-wide score rows (16B align)
static constexpr int MLPD = 132;           // padded ld for 130-wide head-MLP rows

// ======================= device helpers =======================
__device__ __forceinline__ float sigm(float x)   { return 1.f / (1.f + expf(-x)); }
__device__ __forceinline__ float logsig(float x) { return fminf(x, 0.f) - log1pf(expf(-fabsf(x))); }

__device__ __forceinline__ void put4(v16bf& v, int j0, float4 f) {
    v[j0 + 0] = (__bf16)f.x; v[j0 + 1] = (__bf16)f.y;
    v[j0 + 2] = (__bf16)f.z; v[j0 + 3] = (__bf16)f.w;
}

// ======================= generic batched WMMA GEMM =======================
// C[b] = alpha * A[b](MxK) * op(B[b]) (+bias[N]).  A row-major (lda); B row-major
// (ldb) or transposed access (TRANSB: B[n*ldb+k]).  Per-batch offsets:
// off = (b/div)*outer + (b%div)*inner for A, B, C independently.
//
// One wave computes a 16x64 tile: 4 x v8f accumulators, A operand loaded ONCE
// per 32-wide K panel and reused across 4 v_wmma_f32_16x16x32_bf16 issues.
// The fast path issues ALL K-panel loads into independent destinations first
// (so the scheduler can clause them under one counted load-counter wait), then
// converts f32->bf16, then issues the 4 WMMAs.  __launch_bounds__(128, 2)
// trades occupancy for VGPRs so those ~20 b128 destinations can stay live —
// per-wave ILP substitutes for wave count in this latency-bound loop.
// Ragged edges (M=50, N tails, K remainders) fall back to a masked scalar path
// with wave-uniform subtile guards, so EXEC is all-ones at every WMMA
// (ISA 7.12 requirement).
// Alignment contract (guaranteed by all call sites): lda % 4 == 0, and for
// TRANSB also ldb % 4 == 0; base pointers 16B-aligned.
template <int TRANSB>
__global__ void __launch_bounds__(128, 2)
k_gemm_t(const float* __restrict__ A, const float* __restrict__ B,
         const float* __restrict__ bias, float* __restrict__ C,
         int M, int N, int K, int lda, int ldb, int ldc, float alpha,
         long aO, int aD, long aI,
         long bO, int bD, long bI,
         long cO, int cD, long cI)
{
    int b = blockIdx.z;
    const float* Ab = A + (long)(b / aD) * aO + (long)(b % aD) * aI;
    const float* Bb = B + (long)(b / bD) * bO + (long)(b % bD) * bI;
    float*       Cb = C + (long)(b / cD) * cO + (long)(b % cD) * cI;

    int wave = threadIdx.x >> 5;
    int lane = threadIdx.x & 31;
    int m0 = (blockIdx.y * 4 + wave) * 16;
    int n0 = blockIdx.x * 64;
    if (m0 >= M) return;                 // wave-uniform exit
    int g  = lane >> 4;                  // lane half
    int ml = lane & 15;
    int mr = m0 + ml;                    // A row owned by this lane

    v8f acc[4];
#pragma unroll
    for (int u = 0; u < 4; ++u) acc[u] = v8f{0.f,0.f,0.f,0.f,0.f,0.f,0.f,0.f};

    const bool fastAll = (m0 + 16 <= M) && (n0 + 64 <= N);
    int k0 = 0;
    if (fastAll) {
        for (; k0 + 32 <= K; k0 += 32) {
            const float* ar = Ab + (long)mr * lda + k0;
            if (k0 + 32 < K)             // global_prefetch_b8 of next A panel
                __builtin_prefetch(ar + 32, 0, 1);
            // ---- phase 1: issue ALL loads (independent destinations) ----
            // A 16x32 bf16 operand: lane-half g covers K=k0+16g..+15.
            float4 af0 = *(const float4*)(ar + g * 16);
            float4 af1 = *(const float4*)(ar + g * 16 + 4);
            float4 af2 = *(const float4*)(ar + g * 16 + 8);
            float4 af3 = *(const float4*)(ar + g * 16 + 12);
            // B 32x16 bf16 operand per subtile: lane-half g covers K=k0+16g..+15.
            float4 bf[4][4];
            if (TRANSB) {
#pragma unroll
                for (int u = 0; u < 4; ++u) {
                    const float* br = Bb + (long)(n0 + u * 16 + ml) * ldb + k0 + g * 16;
                    bf[u][0] = *(const float4*)(br);
                    bf[u][1] = *(const float4*)(br + 4);
                    bf[u][2] = *(const float4*)(br + 8);
                    bf[u][3] = *(const float4*)(br + 12);
                }
            } else {
                const float* bp = Bb + (long)(k0 + g * 16) * ldb + n0 + ml;
#pragma unroll
                for (int u = 0; u < 4; ++u)
#pragma unroll
                    for (int j = 0; j < 16; ++j)
                        ((float*)&bf[u][0])[j] = bp[(long)j * ldb + u * 16];
            }
            // ---- phase 2: convert everything to bf16 ----
            v16bf av;
            put4(av, 0, af0); put4(av, 4, af1); put4(av, 8, af2); put4(av, 12, af3);
            v16bf bv[4];
#pragma unroll
            for (int u = 0; u < 4; ++u) {
                put4(bv[u], 0, bf[u][0]); put4(bv[u], 4, bf[u][1]);
                put4(bv[u], 8, bf[u][2]); put4(bv[u], 12, bf[u][3]);
            }
            // ---- phase 3: 4 WMMAs reusing the single A operand ----
#pragma unroll
            for (int u = 0; u < 4; ++u)
                acc[u] = __builtin_amdgcn_wmma_f32_16x16x32_bf16(false, av, false, bv[u],
                                                                 (short)0, acc[u], false, false);
        }
    }
    // masked path: K remainder of interior tiles, or whole K for edge tiles
    for (; k0 < K; k0 += 32) {
        v16bf av;
#pragma unroll
        for (int j = 0; j < 16; ++j) {
            int ka = k0 + g * 16 + j;
            av[j] = (__bf16)((mr < M && ka < K) ? Ab[(long)mr * lda + ka] : 0.f);
        }
#pragma unroll
        for (int u = 0; u < 4; ++u) {
            int nb = n0 + u * 16;
            if (nb < N) {                // wave-uniform subtile guard
                int nc = nb + ml;
                v16bf bv;
#pragma unroll
                for (int j = 0; j < 16; ++j) {
                    int kb = k0 + g * 16 + j;
                    float x = 0.f;
                    if (nc < N && kb < K)
                        x = TRANSB ? Bb[(long)nc * ldb + kb] : Bb[(long)kb * ldb + nc];
                    bv[j] = (__bf16)x;
                }
                acc[u] = __builtin_amdgcn_wmma_f32_16x16x32_bf16(false, av, false, bv,
                                                                 (short)0, acc[u], false, false);
            }
        }
    }

    // store: C layout per subtile: lane-half g -> rows 8g..8g+7, col = nb+ml
#pragma unroll
    for (int u = 0; u < 4; ++u) {
        int nb = n0 + u * 16;
        if (nb >= N) continue;           // wave-uniform
        int nc = nb + ml;
        if ((m0 + 16 <= M) && (nb + 16 <= N)) {
            float bb = bias ? bias[nc] : 0.f;
#pragma unroll
            for (int r = 0; r < 8; ++r)
                Cb[(long)(m0 + g * 8 + r) * ldc + nc] = acc[u][r] * alpha + bb;
        } else if (nc < N) {
            float bb = bias ? bias[nc] : 0.f;
#pragma unroll
            for (int r = 0; r < 8; ++r) {
                int mm = m0 + g * 8 + r;
                if (mm < M) Cb[(long)mm * ldc + nc] = acc[u][r] * alpha + bb;
            }
        }
    }
}

// ======================= elementwise / reduction kernels =======================

// LayerNorm over last dim D, one wave per row; optional weight with period
// (per-head weights: widx = (row % period)*D + c). In-place safe.
__global__ void k_ln(const float* __restrict__ x, const float* __restrict__ w,
                     float* __restrict__ y, long rows, int D, float eps, int period)
{
    long row = (long)blockIdx.x * (blockDim.x >> 5) + (threadIdx.x >> 5);
    if (row >= rows) return;
    int lane = threadIdx.x & 31;
    const float* xr = x + row * (long)D;
    float s = 0.f, s2 = 0.f;
    for (int c = lane; c < D; c += 32) { float v = xr[c]; s += v; s2 += v * v; }
    for (int o = 16; o; o >>= 1) { s += __shfl_xor(s, o, 32); s2 += __shfl_xor(s2, o, 32); }
    float mu = s / D;
    float var = s2 / D - mu * mu;
    float r = rsqrtf(var + eps);
    const float* wr = w ? (w + (long)(row % period) * D) : nullptr;
    float* yr = y + row * (long)D;
    for (int c = lane; c < D; c += 32) {
        float v = (xr[c] - mu) * r;
        if (wr) v *= wr[c];
        yr[c] = v;
    }
}

// in-projection mask/split: xf = x*mask, xb = time-reversed x*mask
__global__ void k_in_mask(const float* __restrict__ x, const int* __restrict__ mask,
                          float* __restrict__ xf, float* __restrict__ xb)
{
    long i = (long)blockIdx.x * blockDim.x + threadIdx.x;
    long total = (long)TOKS * DMOD;
    if (i >= total) return;
    int c = (int)(i % DMOD); long rt = i / DMOD;
    int t = (int)(rt % TLEN); int n = (int)(rt / TLEN);
    float v = x[i] * (mask[t * NROW + n] ? 1.f : 0.f);   // mask is (B,T,N,1)
    xf[i] = v;
    xb[((long)n * TLEN + (TLEN - 1 - t)) * DMOD + c] = v;
}

// depthwise causal conv (K=4, left pad 3) + SiLU
__global__ void k_conv_silu(const float* __restrict__ src, int lds,
                            const float* __restrict__ w, const float* __restrict__ b,
                            float* __restrict__ dst, int ldd, int D)
{
    long i = (long)blockIdx.x * blockDim.x + threadIdx.x;
    long total = (long)TOKS * D;
    if (i >= total) return;
    int c = (int)(i % D); long rt = i / D;
    int t = (int)(rt % TLEN); int n = (int)(rt / TLEN);
    const float* xr = src + (long)n * TLEN * lds;
    float acc = b[c];
#pragma unroll
    for (int j = 0; j < 4; ++j) {
        int tt = t + j - 3;
        if (tt >= 0) acc += w[c * 4 + j] * xr[(long)tt * lds + c];
    }
    dst[rt * (long)ldd + c] = acc * sigm(acc);
}

// LinearHeadwiseExpand with 192 heads of 4x4: y[tok,h*4+o] = sum_i x[tok,h*4+i]*w[h,o,i]
__global__ void k_headwise4(const float* __restrict__ src, int lds,
                            const float* __restrict__ w, float* __restrict__ dst)
{
    long i = (long)blockIdx.x * blockDim.x + threadIdx.x;
    long total = (long)TOKS * DINN;
    if (i >= total) return;
    int c = (int)(i % DINN); long tok = i / DINN;
    int h = c >> 2, o = c & 3;
    const float* xr = src + tok * (long)lds + h * 4;
    const float* wr = w + h * 16 + o * 4;
    dst[tok * (long)DINN + c] = xr[0]*wr[0] + xr[1]*wr[1] + xr[2]*wr[2] + xr[3]*wr[3];
}

// ig/fg gate projections over gin = concat(q,k,v) (2304 -> 4)
__global__ void k_gates(const float* __restrict__ q, const float* __restrict__ k,
                        const float* __restrict__ v,
                        const float* __restrict__ igw, const float* __restrict__ igb,
                        const float* __restrict__ fgw, const float* __restrict__ fgb,
                        float* __restrict__ ig, float* __restrict__ fg)
{
    long i = (long)blockIdx.x * blockDim.x + threadIdx.x;
    long total = (long)TOKS * 4;
    if (i >= total) return;
    int g = (int)(i & 3); long tok = i >> 2;
    const float *qr = q + tok * (long)DINN, *kr = k + tok * (long)DINN, *vr = v + tok * (long)DINN;
    float si = igb[g], sf = fgb[g];
    for (int c = 0; c < DINN; ++c) { float x = qr[c]; si += x * igw[c * 4 + g];            sf += x * fgw[c * 4 + g]; }
    for (int c = 0; c < DINN; ++c) { float x = kr[c]; si += x * igw[(DINN + c) * 4 + g];   sf += x * fgw[(DINN + c) * 4 + g]; }
    for (int c = 0; c < DINN; ++c) { float x = vr[c]; si += x * igw[(2*DINN + c) * 4 + g]; sf += x * fgw[(2*DINN + c) * 4 + g]; }
    ig[i] = si; fg[i] = sf;
}

// cumulative log-sigmoid of forget gate per row-head
__global__ void k_cumsum_logsig(const float* __restrict__ fg, float* __restrict__ cs)
{
    int i = blockIdx.x * blockDim.x + threadIdx.x;     // row-head index
    if (i >= BH) return;
    int h = i & 3, r = i >> 2;
    float acc = 0.f;
    for (int t = 0; t < TLEN; ++t) {
        acc += logsig(fg[((long)r * TLEN + t) * 4 + h]);
        cs[(long)i * TLEN + t] = acc;
    }
}

// stabilized mLSTM gating: C = qk*exp(logD-max); normalizer; C /= (norm+1e-6)
__global__ void k_dgate(float* __restrict__ qk, const float* __restrict__ cs,
                        const float* __restrict__ ig)
{
    long i = (long)blockIdx.x * blockDim.x + threadIdx.x;
    long total = (long)BH * TLEN;
    if (i >= total) return;
    int s = (int)(i % TLEN); long b = i / TLEN;
    int h = (int)(b & 3); long r = b >> 2;
    const float* csr = cs + b * TLEN;
    float mx = -3.4e38f;
    for (int t = 0; t <= s; ++t) {
        float ld = csr[s] - csr[t] + ig[((long)r * TLEN + t) * 4 + h];
        mx = fmaxf(mx, ld);
    }
    float* row = qk + b * (long)(TLEN * QKLD) + (long)s * QKLD;
    float sum = 0.f;
    for (int t = 0; t < TLEN; ++t) {
        float cv = 0.f;
        if (t <= s)
            cv = row[t] * expf(csr[s] - csr[t] + ig[((long)r * TLEN + t) * 4 + h] - mx);
        row[t] = cv; sum += cv;
    }
    for (int t = TLEN; t < QKLD; ++t) row[t] = 0.f;     // keep padded K-lanes clean
    float inv = 1.f / (fmaxf(fabsf(sum), expf(-mx)) + 1e-6f);
    for (int t = 0; t <= s; ++t) row[t] *= inv;
}

// mLSTM epilogue: tmp = (hn + skip*xc) * silu(z)   (z = up[:,768:1536])
__global__ void k_post(const float* __restrict__ hn, const float* __restrict__ xc,
                       const float* __restrict__ up, const float* __restrict__ skip,
                       float* __restrict__ out)
{
    long i = (long)blockIdx.x * blockDim.x + threadIdx.x;
    long total = (long)TOKS * DINN;
    if (i >= total) return;
    int c = (int)(i % DINN); long tok = i / DINN;
    float z = up[tok * (long)DUP + DINN + c];
    out[i] = (hn[i] + skip[c] * xc[i]) * (z * sigm(z));
}

__global__ void k_add(float* __restrict__ x, const float* __restrict__ y, long n)
{
    long i = (long)blockIdx.x * blockDim.x + threadIdx.x;
    if (i < n) x[i] += y[i];
}

__global__ void k_relu(float* __restrict__ x, long n)
{
    long i = (long)blockIdx.x * blockDim.x + threadIdx.x;
    if (i < n) x[i] = fmaxf(x[i], 0.f);
}

// GeGLU: act = gelu_exact(up[:,0:512]) * up[:,512:1024]
__global__ void k_geglu(const float* __restrict__ up, float* __restrict__ act)
{
    long i = (long)blockIdx.x * blockDim.x + threadIdx.x;
    long total = (long)TOKS * 512;
    if (i >= total) return;
    int c = (int)(i % 512); long tok = i / 512;
    float g = up[tok * 1024l + c];
    float u = up[tok * 1024l + 512 + c];
    act[i] = 0.5f * g * (1.f + erff(g * 0.70710678118f)) * u;
}

// one sLSTM scan step (elementwise part); rec = h_prev @ R done by k_gemm before
__global__ void k_sstep(const float* __restrict__ gi, const float* __restrict__ gf,
                        const float* __restrict__ gz, const float* __restrict__ go,
                        const float* __restrict__ rec, const float* __restrict__ rb,
                        float* __restrict__ c, float* __restrict__ n,
                        float* __restrict__ h, float* __restrict__ m,
                        float* __restrict__ hs, int t)
{
    long i = (long)blockIdx.x * blockDim.x + threadIdx.x;
    long total = (long)NROW * DMOD;
    if (i >= total) return;
    int col = (int)(i % DMOD); long r = i / DMOD;
    int head = col / DHS, d = col % DHS;
    long tok = r * TLEN + t;
    long gidx = tok * DMOD + col;
    long rbase = (r * NH + head) * (long)(NH * DHS);   // rec (256, head, 4*96)
    float ib = gi[gidx] + rec[rbase + 0 * DHS + d] + rb[(head * 4 + 0) * DHS + d];
    float fb = gf[gidx] + rec[rbase + 1 * DHS + d] + rb[(head * 4 + 1) * DHS + d];
    float zb = gz[gidx] + rec[rbase + 2 * DHS + d] + rb[(head * 4 + 2) * DHS + d];
    float ob = go[gidx] + rec[rbase + 3 * DHS + d] + rb[(head * 4 + 3) * DHS + d];
    float lfm = m[i] + logsig(fb);
    float mn = fmaxf(ib, lfm);
    float ia = expf(ib - mn), fa = expf(lfm - mn);
    float cn = fa * c[i] + ia * tanhf(zb);
    float nn = fa * n[i] + ia;
    float hn = sigm(ob) * cn / nn;
    c[i] = cn; n[i] = nn; h[i] = hn; m[i] = mn;
    hs[gidx] = hn;
}

// feat = ff + time-reversed fb
__global__ void k_feat(const float* __restrict__ xf, const float* __restrict__ xb,
                       float* __restrict__ out)
{
    long i = (long)blockIdx.x * blockDim.x + threadIdx.x;
    long total = (long)TOKS * DMOD;
    if (i >= total) return;
    int c = (int)(i % DMOD); long tok = i / DMOD;
    int t = (int)(tok % TLEN); long n = tok / TLEN;
    out[i] = xf[i] + xb[(n * TLEN + (TLEN - 1 - t)) * (long)DMOD + c];
}

// split flow[...,0:2] / flow[...,2:130] into concatenated flat d_out (ld MLPD src)
__global__ void k_split(const float* __restrict__ m3, float* __restrict__ out)
{
    long i = (long)blockIdx.x * blockDim.x + threadIdx.x;
    long total = (long)TOKS * 130;
    if (i >= total) return;
    int c = (int)(i % 130); long tok = i / 130;
    float v = m3[tok * (long)MLPD + c];
    if (c < 2) out[tok * 2 + c] = v;
    else       out[(long)TOKS * 2 + tok * 128 + (c - 2)] = v;
}

// ======================= host-side orchestration =======================
static inline unsigned eb(long n)  { return (unsigned)((n + 255) / 256); }
static inline unsigned lnb(long r) { return (unsigned)((r + 7) / 8); }   // 8 waves/block

static void gemm(hipStream_t st, const float* A, const float* B, const float* bias, float* C,
                 int M, int N, int K, int lda, int ldb, int ldc, int transB, float alpha,
                 int batch = 1,
                 long aO = 0, int aD = 1, long aI = 0,
                 long bO = 0, int bD = 1, long bI = 0,
                 long cO = 0, int cD = 1, long cI = 0)
{
    dim3 grid((unsigned)((N + 63) / 64), (unsigned)((M + 63) / 64), (unsigned)batch);
    if (transB)
        k_gemm_t<1><<<grid, 128, 0, st>>>(A, B, bias, C, M, N, K, lda, ldb, ldc, alpha,
                                          aO, aD, aI, bO, bD, bI, cO, cD, cI);
    else
        k_gemm_t<0><<<grid, 128, 0, st>>>(A, B, bias, C, M, N, K, lda, ldb, ldc, alpha,
                                          aO, aD, aI, bO, bD, bI, cO, cD, cI);
}

struct MB { const float *ln,*proj_up,*conv_w,*conv_b,*q_w,*k_w,*v_w,
                        *ig_w,*ig_b,*fg_w,*fg_b,*outnorm_w,*skip,*proj_down; };
struct SB { const float *ln,*conv_w,*conv_b,*i_w,*f_w,*z_w,*o_w,
                        *R,*rb,*gn_w,*ffn_ln,*ffn_up,*ffn_down; };
struct Stack { MB m[6]; SB s; const float* post; };

struct Cursor { void* const* in; int i; const float* f() { return (const float*)in[i++]; } };
static void readMB(Cursor& c, MB& m) {
    m.ln=c.f(); m.proj_up=c.f(); m.conv_w=c.f(); m.conv_b=c.f();
    m.q_w=c.f(); m.k_w=c.f(); m.v_w=c.f(); m.ig_w=c.f(); m.ig_b=c.f();
    m.fg_w=c.f(); m.fg_b=c.f(); m.outnorm_w=c.f(); m.skip=c.f(); m.proj_down=c.f();
}
static void readSB(Cursor& c, SB& s) {
    s.ln=c.f(); s.conv_w=c.f(); s.conv_b=c.f(); s.i_w=c.f(); s.f_w=c.f();
    s.z_w=c.f(); s.o_w=c.f(); s.R=c.f(); s.rb=c.f(); s.gn_w=c.f();
    s.ffn_ln=c.f(); s.ffn_up=c.f(); s.ffn_down=c.f();
}
static void readStack(Cursor& c, Stack& st) {
    for (int i = 0; i < 6; ++i) readMB(c, st.m[i]);
    readSB(c, st.s);
    st.post = c.f();
}

struct WS { float *xfwd,*xbwd,*xln,*up,*xc,*qb,*kb,*vb,*hb,*qk,*ig,*fg,*cs,
                  *stc,*stn,*sth,*stm,*rec,*m1,*m2,*m3,*xtmp; };

static void run_mblock(hipStream_t st, const MB& p, float* x, WS& w)
{
    const long TD = (long)TOKS * DMOD;
    k_ln<<<lnb(TOKS), 256, 0, st>>>(x, p.ln, w.xln, TOKS, DMOD, 1e-5f, 1);
    gemm(st, w.xln, p.proj_up, nullptr, w.up, TOKS, DUP, DMOD, DMOD, DUP, DUP, 0, 1.f);
    k_conv_silu<<<eb((long)TOKS*DINN), 256, 0, st>>>(w.up, DUP, p.conv_w, p.conv_b, w.xc, DINN, DINN);
    k_headwise4<<<eb((long)TOKS*DINN), 256, 0, st>>>(w.xc, DINN, p.q_w, w.qb);
    k_headwise4<<<eb((long)TOKS*DINN), 256, 0, st>>>(w.xc, DINN, p.k_w, w.kb);
    k_headwise4<<<eb((long)TOKS*DINN), 256, 0, st>>>(w.up, DUP,  p.v_w, w.vb);
    k_gates<<<eb((long)TOKS*4), 256, 0, st>>>(w.qb, w.kb, w.vb, p.ig_w, p.ig_b, p.fg_w, p.fg_b, w.ig, w.fg);
    k_cumsum_logsig<<<eb(BH), 256, 0, st>>>(w.fg, w.cs);
    // qk[b] = q_b (50x192) @ k_b^T, scale 1/sqrt(DH);  b = r*4 + h  (C padded to ld QKLD)
    gemm(st, w.qb, w.kb, nullptr, w.qk, TLEN, TLEN, DHM, DINN, DINN, QKLD, 1, rsqrtf((float)DHM),
         BH, (long)TLEN*DINN, NH, DHM,  (long)TLEN*DINN, NH, DHM,  (long)TLEN*QKLD, 1, 0);
    k_dgate<<<eb((long)BH*TLEN), 256, 0, st>>>(w.qk, w.cs, w.ig);
    // h[b] = Cn (50x50) @ v_b (50x192)
    gemm(st, w.qk, w.vb, nullptr, w.hb, TLEN, DHM, TLEN, QKLD, DINN, DINN, 0, 1.f,
         BH, (long)TLEN*QKLD, 1, 0,  (long)TLEN*DINN, NH, DHM,  (long)TLEN*DINN, NH, DHM);
    // per-head LN over DH with outnorm weight (period NH)
    k_ln<<<lnb((long)TOKS*NH), 256, 0, st>>>(w.hb, p.outnorm_w, w.hb, (long)TOKS*NH, DHM, 1e-5f, NH);
    k_post<<<eb((long)TOKS*DINN), 256, 0, st>>>(w.hb, w.xc, w.up, p.skip, w.qb);
    gemm(st, w.qb, p.proj_down, nullptr, w.hb, TOKS, DMOD, DINN, DINN, DMOD, DMOD, 0, 1.f);
    k_add<<<eb(TD), 256, 0, st>>>(x, w.hb, TD);
}

static void run_sblock(hipStream_t st, const SB& p, float* x, WS& w)
{
    const long TD = (long)TOKS * DMOD;
    k_ln<<<lnb(TOKS), 256, 0, st>>>(x, p.ln, w.xln, TOKS, DMOD, 1e-5f, 1);
    k_conv_silu<<<eb(TD), 256, 0, st>>>(w.xln, DMOD, p.conv_w, p.conv_b, w.xc, DMOD, DMOD);
    // headwise 4x(96x96) projections as batched WMMA GEMMs (transB: w is (H,o,i))
    const float* srcs[4] = { w.xc, w.xc, w.xln, w.xln };
    const float* wms[4]  = { p.i_w, p.f_w, p.z_w, p.o_w };
    float*       dsts[4] = { w.qb, w.kb, w.vb, w.hb };   // gi, gf, gz, go
    for (int gidx = 0; gidx < 4; ++gidx)
        gemm(st, srcs[gidx], wms[gidx], nullptr, dsts[gidx], TOKS, DHS, DHS, DMOD, DHS, DMOD, 1, 1.f,
             NH, DHS, 1, 0,  (long)DHS*DHS, 1, 0,  DHS, 1, 0);
    size_t sb = (size_t)NROW * DMOD * sizeof(float);
    hipMemsetAsync(w.stc, 0, sb, st);
    hipMemsetAsync(w.stn, 0, sb, st);
    hipMemsetAsync(w.sth, 0, sb, st);
    hipMemsetAsync(w.stm, 0, sb, st);
    for (int t = 0; t < TLEN; ++t) {
        // rec[b,h,:] = h_prev[b, h*96:...] @ R[h] (96 x 384); batched over heads
        gemm(st, w.sth, p.R, nullptr, w.rec, NROW, NH*DHS, DHS, DMOD, NH*DHS, NH*NH*DHS, 0, 1.f,
             NH, DHS, 1, 0,  (long)DHS*NH*DHS, 1, 0,  (long)NH*DHS, 1, 0);
        k_sstep<<<eb((long)NROW*DMOD), 256, 0, st>>>(w.qb, w.kb, w.vb, w.hb, w.rec, p.rb,
                                                     w.stc, w.stn, w.sth, w.stm, w.up, t);
    }
    k_ln<<<lnb((long)TOKS*NH), 256, 0, st>>>(w.up, p.gn_w, w.xln, (long)TOKS*NH, DHS, 1e-5f, NH);
    k_add<<<eb(TD), 256, 0, st>>>(x, w.xln, TD);
    // FFN (GeGLU)
    k_ln<<<lnb(TOKS), 256, 0, st>>>(x, p.ffn_ln, w.xln, TOKS, DMOD, 1e-5f, 1);
    gemm(st, w.xln, p.ffn_up, nullptr, w.up, TOKS, 1024, DMOD, DMOD, 1024, 1024, 0, 1.f);
    k_geglu<<<eb((long)TOKS*512), 256, 0, st>>>(w.up, w.xc);
    gemm(st, w.xc, p.ffn_down, nullptr, w.qb, TOKS, DMOD, 512, 512, DMOD, DMOD, 0, 1.f);
    k_add<<<eb(TD), 256, 0, st>>>(x, w.qb, TD);
}

static void run_stack(hipStream_t st, const Stack& S, float* x, WS& w)
{
    // block order: m0, s, m1, m2, m3, m4, m5
    run_mblock(st, S.m[0], x, w);
    run_sblock(st, S.s,    x, w);
    for (int i = 1; i < 6; ++i) run_mblock(st, S.m[i], x, w);
    k_ln<<<lnb(TOKS), 256, 0, st>>>(x, S.post, x, TOKS, DMOD, 1e-5f, 1);
}

extern "C" void kernel_launch(void* const* d_in, const int* in_sizes, int n_in,
                              void* d_out, int out_size, void* d_ws, size_t ws_size,
                              hipStream_t stream)
{
    (void)in_sizes; (void)n_in; (void)out_size; (void)ws_size;

    // ---- unpack inputs (setup_inputs dict insertion order) ----
    const float* input = (const float*)d_in[0];      // (1,256,50,320)
    const int*   mask  = (const int*)d_in[1];        // (1,50,256,1)
    Cursor c{d_in, 2};
    const float* in_w = c.f();
    const float* in_b = c.f();
    Stack fwd, bwd;
    readStack(c, fwd);
    readStack(c, bwd);
    const float* fh_w1 = c.f(); const float* fh_b1 = c.f();
    const float* fh_w2 = c.f(); const float* fh_b2 = c.f();
    const float* fh_w3 = c.f(); const float* fh_b3 = c.f();

    // ---- carve workspace ----
    float* p = (float*)d_ws;
    auto take = [&](long n) { float* r = p; p += n; return r; };
    WS w;
    w.xfwd = take((long)TOKS * DMOD);
    w.xbwd = take((long)TOKS * DMOD);
    w.xln  = take((long)TOKS * DMOD);
    w.up   = take((long)TOKS * DUP);
    w.xc   = take((long)TOKS * DINN);
    w.qb   = take((long)TOKS * DINN);
    w.kb   = take((long)TOKS * DINN);
    w.vb   = take((long)TOKS * DINN);
    w.hb   = take((long)TOKS * DINN);
    w.qk   = take((long)BH * TLEN * QKLD);
    w.ig   = take((long)TOKS * 4);
    w.fg   = take((long)TOKS * 4);
    w.cs   = take((long)BH * TLEN);
    w.stc  = take((long)NROW * DMOD);
    w.stn  = take((long)NROW * DMOD);
    w.sth  = take((long)NROW * DMOD);
    w.stm  = take((long)NROW * DMOD);
    w.rec  = take((long)NROW * NH * NH * DHS);
    w.m1   = take((long)TOKS * MLPD);
    w.m2   = take((long)TOKS * MLPD);
    w.m3   = take((long)TOKS * MLPD);
    w.xtmp = take((long)TOKS * DMOD);

    // ---- input projection + mask + direction split + pre-stack LN (eps 1e-6) ----
    gemm(stream, input, in_w, in_b, w.xtmp, TOKS, DMOD, 320, 320, DMOD, DMOD, 0, 1.f);
    k_in_mask<<<eb((long)TOKS*DMOD), 256, 0, stream>>>(w.xtmp, mask, w.xfwd, w.xbwd);
    k_ln<<<lnb(TOKS), 256, 0, stream>>>(w.xfwd, nullptr, w.xfwd, TOKS, DMOD, 1e-6f, 1);
    k_ln<<<lnb(TOKS), 256, 0, stream>>>(w.xbwd, nullptr, w.xbwd, TOKS, DMOD, 1e-6f, 1);

    // ---- bidirectional stacks ----
    run_stack(stream, fwd, w.xfwd, w);
    run_stack(stream, bwd, w.xbwd, w);

    // ---- combine + head MLP (padded ld MLPD; K-bounds mask the pad columns) ----
    k_feat<<<eb((long)TOKS*DMOD), 256, 0, stream>>>(w.xfwd, w.xbwd, w.xln);
    gemm(stream, w.xln, fh_w1, fh_b1, w.m1, TOKS, 130, DMOD, DMOD, 130, MLPD, 0, 1.f);
    k_relu<<<eb((long)TOKS*MLPD), 256, 0, stream>>>(w.m1, (long)TOKS*MLPD);
    gemm(stream, w.m1, fh_w2, fh_b2, w.m2, TOKS, 130, 130, MLPD, 130, MLPD, 0, 1.f);
    k_relu<<<eb((long)TOKS*MLPD), 256, 0, stream>>>(w.m2, (long)TOKS*MLPD);
    gemm(stream, w.m2, fh_w3, fh_b3, w.m3, TOKS, 130, 130, MLPD, 130, MLPD, 0, 1.f);
    k_split<<<eb((long)TOKS*130), 256, 0, stream>>>(w.m3, (float*)d_out);
}
